// Block_ViT_29961691856979
// MI455X (gfx1250) — matrix operand order repaired and checked by reading, not compiled, not run on hardware
//
#include <hip/hip_runtime.h>
#include <hip/hip_bf16.h>
#include <math.h>

// ---------------------------------------------------------------------------
// Types for WMMA fragments (CDNA5 / gfx1250, wave32)
// ---------------------------------------------------------------------------
typedef __attribute__((ext_vector_type(16))) _Float16 v16h;
typedef __attribute__((ext_vector_type(8)))  _Float16 h8;
typedef __attribute__((ext_vector_type(8)))  float    v8f;

union Frag { v16h v; h8 h[2]; };

#define TM 64
#define TN 64
#define TK 32
#define LDP (TK + 8)   // padded LDS row stride (halfwords), 80B = bank-friendly

// ---------------------------------------------------------------------------
// Batched GEMM, fp32 in/out, f16 WMMA math with f32 accumulation.
//   C[b] (M x N) = alpha * A[b] (M x K, row-major) * B[b] (K x N)
//   transB==0 : B stored row-major K x N  (element (k,n) = B[k*N + n])
//   transB==1 : B stored row-major N x K  (element (k,n) = B[n*K + k])
//   strides sA/sB/sC may be 0 to share an operand across the batch.
// Requires: M % 64 == 0, N % 64 == 0, K % 32 == 0 (true for all calls here).
//
// Software pipelined: global loads for tile k+1 are issued before the WMMAs
// of tile k; LDS is double-buffered so one barrier per iteration suffices.
// ---------------------------------------------------------------------------
__global__ void __launch_bounds__(128)
gemm_wmma_f16(const float* __restrict__ A, const float* __restrict__ Bm,
              float* __restrict__ C,
              int M, int N, int K,
              long long sA, long long sB, long long sC,
              int transB, float alpha)
{
  __shared__ __align__(16) _Float16 Asub[2][TM][LDP];
  __shared__ __align__(16) _Float16 Bsub[2][TN][LDP];

  const int b  = blockIdx.z;
  const int Mt = blockIdx.y * TM;
  const int Nt = blockIdx.x * TN;
  const float* Ab = A  + (long long)b * sA;
  const float* Bb = Bm + (long long)b * sB;
  float*       Cb = C  + (long long)b * sC;

  const int tid  = threadIdx.x;
  const int wid  = tid >> 5;
  const int lane = tid & 31;
  const int l16  = lane & 15;
  const int hl   = lane >> 4;         // which 16-lane half of the wave
  const int wm   = (wid >> 1) * 32;   // wave row offset inside tile
  const int wn   = (wid & 1) * 32;    // wave col offset inside tile

  // staging indices (constant per thread)
  int ar[4], ac4[4], bk[4], bn4[4];
  for (int it = 0; it < 4; ++it) {
    int idx = tid + it * 128;   // 0..511 float4 chunks
    ar[it]  = idx >> 3;         // 0..63 rows (A, and B when transposed)
    ac4[it] = (idx & 7) * 4;    // k offset 0..28
    bk[it]  = idx >> 4;         // 0..31 k rows (B row-major)
    bn4[it] = (idx & 15) * 4;   // n offset 0..60
  }

  v8f acc[2][2] = {};
  float4 ra[4], rb[4];
  const int nIter = K / TK;

  // ---- issue + stage tile 0 ----
  for (int it = 0; it < 4; ++it)
    ra[it] = *(const float4*)(Ab + (long long)(Mt + ar[it]) * K + ac4[it]);
  if (transB) {
    for (int it = 0; it < 4; ++it)
      rb[it] = *(const float4*)(Bb + (long long)(Nt + ar[it]) * K + ac4[it]);
  } else {
    for (int it = 0; it < 4; ++it)
      rb[it] = *(const float4*)(Bb + (long long)bk[it] * N + Nt + bn4[it]);
  }
  for (int it = 0; it < 4; ++it) {
    Asub[0][ar[it]][ac4[it] + 0] = (_Float16)ra[it].x;
    Asub[0][ar[it]][ac4[it] + 1] = (_Float16)ra[it].y;
    Asub[0][ar[it]][ac4[it] + 2] = (_Float16)ra[it].z;
    Asub[0][ar[it]][ac4[it] + 3] = (_Float16)ra[it].w;
  }
  if (transB) {
    for (int it = 0; it < 4; ++it) {
      Bsub[0][ar[it]][ac4[it] + 0] = (_Float16)rb[it].x;
      Bsub[0][ar[it]][ac4[it] + 1] = (_Float16)rb[it].y;
      Bsub[0][ar[it]][ac4[it] + 2] = (_Float16)rb[it].z;
      Bsub[0][ar[it]][ac4[it] + 3] = (_Float16)rb[it].w;
    }
  } else {
    for (int it = 0; it < 4; ++it) {
      Bsub[0][bn4[it] + 0][bk[it]] = (_Float16)rb[it].x;
      Bsub[0][bn4[it] + 1][bk[it]] = (_Float16)rb[it].y;
      Bsub[0][bn4[it] + 2][bk[it]] = (_Float16)rb[it].z;
      Bsub[0][bn4[it] + 3][bk[it]] = (_Float16)rb[it].w;
    }
  }
  __syncthreads();

  for (int iter = 0; iter < nIter; ++iter) {
    const int p = iter & 1;
    const bool hasNext = (iter + 1) < nIter;

    // ---- issue global loads for next tile (hide latency under WMMAs) ----
    if (hasNext) {
      const int k0 = (iter + 1) * TK;
      for (int it = 0; it < 4; ++it)
        ra[it] = *(const float4*)(Ab + (long long)(Mt + ar[it]) * K + k0 + ac4[it]);
      if (transB) {
        for (int it = 0; it < 4; ++it)
          rb[it] = *(const float4*)(Bb + (long long)(Nt + ar[it]) * K + k0 + ac4[it]);
      } else {
        for (int it = 0; it < 4; ++it)
          rb[it] = *(const float4*)(Bb + (long long)(k0 + bk[it]) * N + Nt + bn4[it]);
      }
    }

    // ---- fragments per ISA VGPR layouts + 2x2 WMMA ----
    // A (16x32 f16): lane half 0 holds K {0..7,16..23}, half 1 {8..15,24..31}
    Frag a[2], bf[2];
    for (int m = 0; m < 2; ++m) {
      const _Float16* ap = &Asub[p][wm + m * 16 + l16][0];
      a[m].h[0] = *(const h8*)(ap + 8 * hl);
      a[m].h[1] = *(const h8*)(ap + 16 + 8 * hl);
    }
    // B (32x16 f16): lane half 0 holds K {0..7,16..23}, half 1 {8..15,24..31}
    for (int n = 0; n < 2; ++n) {
      const _Float16* bp = &Bsub[p][wn + n * 16 + l16][0];
      bf[n].h[0] = *(const h8*)(bp + 8 * hl);
      bf[n].h[1] = *(const h8*)(bp + 16 + 8 * hl);
    }
    for (int m = 0; m < 2; ++m)
      for (int n = 0; n < 2; ++n)
        acc[m][n] = __builtin_amdgcn_wmma_f32_16x16x32_f16(
            false, a[m].v, false, bf[n].v, (short)0, acc[m][n], false, false);

    // ---- convert + store next tile into the other LDS buffer ----
    if (hasNext) {
      const int q = p ^ 1;
      for (int it = 0; it < 4; ++it) {
        Asub[q][ar[it]][ac4[it] + 0] = (_Float16)ra[it].x;
        Asub[q][ar[it]][ac4[it] + 1] = (_Float16)ra[it].y;
        Asub[q][ar[it]][ac4[it] + 2] = (_Float16)ra[it].z;
        Asub[q][ar[it]][ac4[it] + 3] = (_Float16)ra[it].w;
      }
      if (transB) {
        for (int it = 0; it < 4; ++it) {
          Bsub[q][ar[it]][ac4[it] + 0] = (_Float16)rb[it].x;
          Bsub[q][ar[it]][ac4[it] + 1] = (_Float16)rb[it].y;
          Bsub[q][ar[it]][ac4[it] + 2] = (_Float16)rb[it].z;
          Bsub[q][ar[it]][ac4[it] + 3] = (_Float16)rb[it].w;
        }
      } else {
        for (int it = 0; it < 4; ++it) {
          Bsub[q][bn4[it] + 0][bk[it]] = (_Float16)rb[it].x;
          Bsub[q][bn4[it] + 1][bk[it]] = (_Float16)rb[it].y;
          Bsub[q][bn4[it] + 2][bk[it]] = (_Float16)rb[it].z;
          Bsub[q][bn4[it] + 3][bk[it]] = (_Float16)rb[it].w;
        }
      }
    }
    __syncthreads();
  }

  // ---- epilogue: C/D layout = lane col l16, rows v + 8*hl ----
  for (int m = 0; m < 2; ++m)
    for (int n = 0; n < 2; ++n) {
      int col = Nt + wn + n * 16 + l16;
      for (int v = 0; v < 8; ++v) {
        int row = Mt + wm + m * 16 + v + 8 * hl;
        Cb[(long long)row * N + col] = alpha * acc[m][n][v];
      }
    }
}

// ---------------------------------------------------------------------------
// Channel concat copy: dst[b, c0+c, p] = src[b, c, p]
// ---------------------------------------------------------------------------
__global__ void chan_copy(const float* __restrict__ src, float* __restrict__ dst,
                          int C, int Ctot, int c0, long long HW, long long n)
{
  long long i = blockIdx.x * (long long)blockDim.x + threadIdx.x;
  if (i >= n) return;
  long long p = i % HW;
  long long t = i / HW;
  int c = (int)(t % C);
  int b = (int)(t / C);
  dst[((long long)b * Ctot + c0 + c) * HW + p] = src[i];
}

// ---------------------------------------------------------------------------
// LayerNorm over channel dim per pixel (thread per pixel, coalesced in p)
// ---------------------------------------------------------------------------
__global__ void ln_c_kernel(const float* __restrict__ x, const float* __restrict__ w,
                            const float* __restrict__ bb, float* __restrict__ y,
                            int C, int HW, int npix)
{
  int i = blockIdx.x * blockDim.x + threadIdx.x;
  if (i >= npix) return;
  int b = i / HW, p = i % HW;
  const float* xp = x + (long long)b * C * HW + p;
  float s = 0.f, s2 = 0.f;
  for (int c = 0; c < C; ++c) {
    float v = xp[(long long)c * HW];
    s += v; s2 += v * v;
  }
  float mu  = s / C;
  float var = s2 / C - mu * mu;
  float rs  = rsqrtf(var + 1e-5f);
  float* yp = y + (long long)b * C * HW + p;
  for (int c = 0; c < C; ++c)
    yp[(long long)c * HW] = (xp[(long long)c * HW] - mu) * rs * w[c] + bb[c];
}

// ---------------------------------------------------------------------------
// Grouped conv (covers depthwise 3x3/5x5 and ipg=2/opg=2), optional ReLU.
// ---------------------------------------------------------------------------
__global__ void gconv_kernel(const float* __restrict__ in, const float* __restrict__ wt,
                             float* __restrict__ out, int Cout, int ipg, int opg,
                             int H, int W, int ks, int pad, int relu,
                             int inCtot, int inCoff, int outCtot, int outCoff,
                             long long n)
{
  long long i = blockIdx.x * (long long)blockDim.x + threadIdx.x;
  if (i >= n) return;
  int x = (int)(i % W);
  long long t = i / W;
  int y = (int)(t % H); t /= H;
  int o = (int)(t % Cout);
  int b = (int)(t / Cout);
  int g = o / opg;
  const float* wp = wt + (long long)o * ipg * ks * ks;
  float acc = 0.f;
  for (int ci = 0; ci < ipg; ++ci) {
    const float* ip = in + ((long long)b * inCtot + inCoff + g * ipg + ci) * H * W;
    for (int dy = 0; dy < ks; ++dy) {
      int yy = y + dy - pad;
      if (yy < 0 || yy >= H) continue;
      for (int dx = 0; dx < ks; ++dx) {
        int xx = x + dx - pad;
        if (xx < 0 || xx >= W) continue;
        acc += wp[(ci * ks + dy) * ks + dx] * ip[yy * W + xx];
      }
    }
  }
  if (relu) acc = fmaxf(acc, 0.f);
  out[((long long)b * outCtot + outCoff + o) * H * W + (long long)y * W + x] = acc;
}

// ---------------------------------------------------------------------------
// In-place L2-normalize rows of length L (wave per row)
// ---------------------------------------------------------------------------
__global__ void l2norm_rows(float* __restrict__ x, int rows, int L)
{
  int r    = (blockIdx.x * blockDim.x + threadIdx.x) >> 5;
  int lane = threadIdx.x & 31;
  if (r >= rows) return;
  float* p = x + (long long)r * L;
  float s = 0.f;
  for (int i = lane; i < L; i += 32) { float v = p[i]; s += v * v; }
  for (int o = 16; o; o >>= 1) s += __shfl_xor(s, o, 32);
  float inv = 1.f / fmaxf(sqrtf(s), 1e-12f);
  for (int i = lane; i < L; i += 32) p[i] *= inv;
}

// ---------------------------------------------------------------------------
// Per-batch mean/var over Ntot elements (one block per batch, deterministic)
// ---------------------------------------------------------------------------
__global__ void mean_var_kernel(const float* __restrict__ x, float* __restrict__ stats,
                                long long Ntot)
{
  int b = blockIdx.x;
  const float* p = x + (long long)b * Ntot;
  float s = 0.f, s2 = 0.f;
  for (long long i = threadIdx.x; i < Ntot; i += blockDim.x) {
    float v = p[i]; s += v; s2 += v * v;
  }
  __shared__ float sh[256], sh2[256];
  sh[threadIdx.x] = s; sh2[threadIdx.x] = s2;
  __syncthreads();
  for (int off = 128; off; off >>= 1) {
    if (threadIdx.x < off) { sh[threadIdx.x] += sh[threadIdx.x + off];
                             sh2[threadIdx.x] += sh2[threadIdx.x + off]; }
    __syncthreads();
  }
  if (threadIdx.x == 0) {
    float mu = sh[0] / (float)Ntot;
    stats[b * 2 + 0] = mu;
    stats[b * 2 + 1] = sh2[0] / (float)Ntot - mu * mu;
  }
}

// ---------------------------------------------------------------------------
// inorm (whole-matrix per batch) + softmax along row (wave per row, in place)
// ---------------------------------------------------------------------------
__global__ void inorm_softmax(float* __restrict__ s, const float* __restrict__ stats,
                              int rowsPerB, int Ncol, int totalRows)
{
  int r    = (blockIdx.x * blockDim.x + threadIdx.x) >> 5;
  int lane = threadIdx.x & 31;
  if (r >= totalRows) return;
  int b = r / rowsPerB;
  float mu = stats[b * 2 + 0];
  float rs = rsqrtf(stats[b * 2 + 1] + 1e-5f);
  float* row = s + (long long)r * Ncol;
  float mx = -3.4e38f;
  for (int i = lane; i < Ncol; i += 32) mx = fmaxf(mx, (row[i] - mu) * rs);
  for (int o = 16; o; o >>= 1) mx = fmaxf(mx, __shfl_xor(mx, o, 32));
  float sum = 0.f;
  for (int i = lane; i < Ncol; i += 32) sum += __expf((row[i] - mu) * rs - mx);
  for (int o = 16; o; o >>= 1) sum += __shfl_xor(sum, o, 32);
  float inv = 1.f / sum;
  for (int i = lane; i < Ncol; i += 32)
    row[i] = __expf((row[i] - mu) * rs - mx) * inv;
}

// ---------------------------------------------------------------------------
// Elementwise residual add
// ---------------------------------------------------------------------------
__global__ void add_kernel(const float* __restrict__ a, const float* __restrict__ b,
                           float* __restrict__ o, long long n)
{
  long long i = blockIdx.x * (long long)blockDim.x + threadIdx.x;
  if (i >= n) return;
  o[i] = a[i] + b[i];
}

// ---------------------------------------------------------------------------
// Mean over row of length L (wave per row) -> outv[row]
// ---------------------------------------------------------------------------
__global__ void rowmean_kernel(const float* __restrict__ x, float* __restrict__ outv,
                               int rows, int L)
{
  int r    = (blockIdx.x * blockDim.x + threadIdx.x) >> 5;
  int lane = threadIdx.x & 31;
  if (r >= rows) return;
  const float* p = x + (long long)r * L;
  float s = 0.f;
  for (int i = lane; i < L; i += 32) s += p[i];
  for (int o = 16; o; o >>= 1) s += __shfl_xor(s, o, 32);
  if (lane == 0) outv[r] = s / (float)L;
}

// ---------------------------------------------------------------------------
// ECA (1D conv k=3 over channel of pooled vec) + sigmoid gate + residual
// ---------------------------------------------------------------------------
__global__ void eca_gate_add(const float* __restrict__ y, const float* __restrict__ pool,
                             const float* __restrict__ eca, const float* __restrict__ res,
                             float* __restrict__ o, int C, long long HW, long long n)
{
  long long i = blockIdx.x * (long long)blockDim.x + threadIdx.x;
  if (i >= n) return;
  long long t = i / HW;
  int c = (int)(t % C);
  int b = (int)(t / C);
  const float* pb = pool + (long long)b * C;
  float a = pb[c] * eca[1];
  if (c > 0)     a += pb[c - 1] * eca[0];
  if (c < C - 1) a += pb[c + 1] * eca[2];
  float g = 1.f / (1.f + __expf(-a));
  o[i] = y[i] * g + res[i];
}

// ---------------------------------------------------------------------------
// Host orchestration
// ---------------------------------------------------------------------------
static inline unsigned cdivu(long long a, long long b) {
  return (unsigned)((a + b - 1) / b);
}

extern "C" void kernel_launch(void* const* d_in, const int* in_sizes, int n_in,
                              void* d_out, int out_size, void* d_ws, size_t ws_size,
                              hipStream_t stream)
{
  (void)in_sizes; (void)n_in; (void)out_size; (void)ws_size;

  const int  Bn = 4, Hh = 56, Ww = 56;
  const long long HW = (long long)Hh * Ww;   // 3136
  const int  KVc = 960;
  const int  CHs[4] = {64, 128, 256, 512};
  const float scale = 0.032274861218395f;    // 1/sqrt(960)

  // --- input pointers per setup_inputs() dict (insertion) order ---
  const float *anw[4], *anb[4], *fnw[4], *fnb[4];
  for (int i = 0; i < 4; ++i) {
    anw[i] = (const float*)d_in[4 * i + 0];
    anb[i] = (const float*)d_in[4 * i + 1];
    fnw[i] = (const float*)d_in[4 * i + 2];
    fnb[i] = (const float*)d_in[4 * i + 3];
  }
  const float* an_all_w = (const float*)d_in[16];
  const float* an_all_b = (const float*)d_in[17];
  const float* mheadq_C = (const float*)d_in[18];
  const float* mheadk_C = (const float*)d_in[19];
  const float* mheadv_C = (const float*)d_in[20];
  const float* q_C = (const float*)d_in[21];
  const float* k_C = (const float*)d_in[22];
  const float* v_C = (const float*)d_in[23];
  const float *mhead[4], *qw[4], *proj[4];
  for (int i = 0; i < 4; ++i) {
    mhead[i] = (const float*)d_in[24 + 3 * i];
    qw[i]    = (const float*)d_in[25 + 3 * i];
    proj[i]  = (const float*)d_in[26 + 3 * i];
  }
  const float *pin[4], *dw3[4], *dw5[4], *pout[4], *eca[4];
  for (int i = 0; i < 4; ++i) {
    pin[i]  = (const float*)d_in[36 + 5 * i];
    dw3[i]  = (const float*)d_in[37 + 5 * i];
    dw5[i]  = (const float*)d_in[38 + 5 * i];
    pout[i] = (const float*)d_in[39 + 5 * i];
    eca[i]  = (const float*)d_in[40 + 5 * i];
  }
  const float* emb[4] = {(const float*)d_in[56], (const float*)d_in[57],
                         (const float*)d_in[58], (const float*)d_in[59]};

  // --- workspace layout (floats), regions reused across stages ---
  float* ws = (float*)d_ws;
  const long long S_all = (long long)Bn * KVc * HW;   // 12,042,240
  float* CTX  = ws;                 // emb_all, later ctx
  float* BUFA = ws + 1 * S_all;     // embcat, later K_n; per-scale P-region base
  float* BUFC = ws + 2 * S_all;     // QKV pre-dw, later sim
  float* BUFD = ws + 3 * S_all;     // Q_n
  float* BUFE = ws + 4 * S_all;     // V
  const long long Pc = (long long)Bn * 512 * HW;      // 6,422,528
  float* P0  = BUFA;
  float* P1  = P0 + Pc;
  float* P2  = P1 + Pc;                                // up to B*512*KV
  float* P3  = P2 + (long long)Bn * 512 * KVc;
  float* FB1 = P3 + Pc;                                // up to B*8c*HW
  float* FB2 = FB1 + 8 * Pc;
  float* STATS = FB2 + 8 * Pc;                         // 8 floats used
  float* POOL  = STATS + 16;                           // up to B*512

  auto gemm = [&](const float* A, long long sA, const float* Bm, long long sB,
                  float* Cc, long long sC, int M, int N, int K, int transB,
                  float alpha) {
    dim3 g((unsigned)(N / 64), (unsigned)(M / 64), (unsigned)Bn);
    gemm_wmma_f16<<<g, dim3(128), 0, stream>>>(A, Bm, Cc, M, N, K, sA, sB, sC,
                                               transB, alpha);
  };

  const int npix = (int)(Bn * HW);
  const long long Skv = (long long)KVc * HW;

  // ---- embcat + LayerNorm over all 960 channels -> emb_all (CTX) ----
  {
    int c0 = 0;
    for (int i = 0; i < 4; ++i) {
      long long n = (long long)Bn * CHs[i] * HW;
      chan_copy<<<cdivu(n, 256), 256, 0, stream>>>(emb[i], BUFA, CHs[i], KVc, c0, HW, n);
      c0 += CHs[i];
    }
    ln_c_kernel<<<cdivu(npix, 256), 256, 0, stream>>>(BUFA, an_all_w, an_all_b,
                                                      CTX, KVc, (int)HW, npix);
  }

  // ---- Q/K/V: conv1x1 (WMMA GEMM) -> depthwise 3x3 -> (l2norm for Q,K) ----
  const long long nAll = (long long)Bn * KVc * HW;
  // Q
  gemm(mheadq_C, 0, CTX, Skv, BUFC, Skv, KVc, (int)HW, KVc, 0, 1.f);
  gconv_kernel<<<cdivu(nAll, 256), 256, 0, stream>>>(BUFC, q_C, BUFD, KVc, 1, 1,
      Hh, Ww, 3, 1, 0, KVc, 0, KVc, 0, nAll);
  l2norm_rows<<<cdivu((long long)Bn * KVc * 32, 256), 256, 0, stream>>>(BUFD, Bn * KVc, (int)HW);
  // K
  gemm(mheadk_C, 0, CTX, Skv, BUFC, Skv, KVc, (int)HW, KVc, 0, 1.f);
  gconv_kernel<<<cdivu(nAll, 256), 256, 0, stream>>>(BUFC, k_C, BUFA, KVc, 1, 1,
      Hh, Ww, 3, 1, 0, KVc, 0, KVc, 0, nAll);
  l2norm_rows<<<cdivu((long long)Bn * KVc * 32, 256), 256, 0, stream>>>(BUFA, Bn * KVc, (int)HW);
  // V
  gemm(mheadv_C, 0, CTX, Skv, BUFC, Skv, KVc, (int)HW, KVc, 0, 1.f);
  gconv_kernel<<<cdivu(nAll, 256), 256, 0, stream>>>(BUFC, v_C, BUFE, KVc, 1, 1,
      Hh, Ww, 3, 1, 0, KVc, 0, KVc, 0, nAll);

  // ---- sim = scale * Q_n . K_n^T ; inorm+softmax ; ctx = sim . V ----
  gemm(BUFD, Skv, BUFA, Skv, BUFC, (long long)KVc * KVc, KVc, KVc, (int)HW, 1, scale);
  mean_var_kernel<<<Bn, 256, 0, stream>>>(BUFC, STATS, (long long)KVc * KVc);
  inorm_softmax<<<cdivu((long long)Bn * KVc * 32, 256), 256, 0, stream>>>(
      BUFC, STATS, KVc, KVc, Bn * KVc);
  gemm(BUFC, (long long)KVc * KVc, BUFE, Skv, CTX, Skv, KVc, (int)HW, KVc, 0, 1.f);
  // CTX now holds ctx (B, KV, HW); everything else is free.

  // ---- per-scale attention read-out + FFN ----
  long long out_off = 0;
  for (int i = 0; i < 4; ++i) {
    const int c  = CHs[i];
    const int hf = 4 * c;
    const int c2 = 8 * c;
    const long long nC  = (long long)Bn * c * HW;
    const long long nHf = (long long)Bn * hf * HW;
    const long long Sc  = (long long)c * HW;

    // cx = LN(emb_i) ; q = gconv(conv1x1(cx)) ; l2norm over hw
    ln_c_kernel<<<cdivu(npix, 256), 256, 0, stream>>>(emb[i], anw[i], anb[i], P0, c, (int)HW, npix);
    gemm(mhead[i], 0, P0, Sc, P1, Sc, c, (int)HW, c, 0, 1.f);
    gconv_kernel<<<cdivu(nC, 256), 256, 0, stream>>>(P1, qw[i], P0, c, 2, 2,
        Hh, Ww, 3, 1, 0, c, 0, c, 0, nC);
    l2norm_rows<<<cdivu((long long)Bn * c * 32, 256), 256, 0, stream>>>(P0, Bn * c, (int)HW);

    // attn = softmax(inorm(scale * q . ctx^T)) ; o = attn . ctx
    gemm(P0, Sc, CTX, Skv, P2, (long long)c * KVc, c, KVc, (int)HW, 1, scale);
    mean_var_kernel<<<Bn, 256, 0, stream>>>(P2, STATS, (long long)c * KVc);
    inorm_softmax<<<cdivu((long long)Bn * c * 32, 256), 256, 0, stream>>>(
        P2, STATS, c, KVc, Bn * c);
    gemm(P2, (long long)c * KVc, CTX, Skv, P1, Sc, c, (int)HW, KVc, 0, 1.f);

    // project_out + residual
    gemm(proj[i], 0, P1, Sc, P0, Sc, c, (int)HW, c, 0, 1.f);
    add_kernel<<<cdivu(nC, 256), 256, 0, stream>>>(emb[i], P0, P3, nC);

    // FFN: LN -> pin -> dw3/dw5 + relu -> pout -> ECA gate, + residual
    ln_c_kernel<<<cdivu(npix, 256), 256, 0, stream>>>(P3, fnw[i], fnb[i], P0, c, (int)HW, npix);
    gemm(pin[i], 0, P0, Sc, FB1, (long long)c2 * HW, c2, (int)HW, c, 0, 1.f);
    gconv_kernel<<<cdivu(nHf, 256), 256, 0, stream>>>(FB1, dw3[i], FB2, hf, 1, 1,
        Hh, Ww, 3, 1, 1, c2, 0, c2, 0, nHf);
    gconv_kernel<<<cdivu(nHf, 256), 256, 0, stream>>>(FB1, dw5[i], FB2, hf, 1, 1,
        Hh, Ww, 5, 2, 1, c2, hf, c2, hf, nHf);
    gemm(pout[i], 0, FB2, (long long)c2 * HW, P0, Sc, c, (int)HW, c2, 0, 1.f);
    rowmean_kernel<<<cdivu((long long)Bn * c * 32, 256), 256, 0, stream>>>(P0, POOL, Bn * c, (int)HW);
    eca_gate_add<<<cdivu(nC, 256), 256, 0, stream>>>(P0, POOL, eca[i], P3,
        (float*)d_out + out_off, c, HW, nC);
    out_off += nC;
  }
}
